// CA_Module_50646254355183
// MI455X (gfx1250) — matrix operand order repaired and checked by reading, not compiled, not run on hardware
//
#include <hip/hip_runtime.h>
#include <hip/hip_bf16.h>

typedef __bf16 bf16_t;
typedef __attribute__((ext_vector_type(16))) __bf16 v16bf;
typedef __attribute__((ext_vector_type(8)))  float  v8f;

#define BB 16
#define CC 512
#define NN 4096

union FragU { uint4 u[2]; v16bf v; };

// A fragment: 16x32 bf16, row-major source with stride ld.
// Layout used here (contiguous k, not the ISA table order): lane L -> row (L&15); lanes 0-15 hold K 0..15 contiguous, lanes 16-31 hold K 16..31.
__device__ inline v16bf loadFragA(const bf16_t* __restrict__ base, int ld, int k, int lane) {
    const int row = lane & 15;
    const int ko  = (lane >> 4) * 16;
    const bf16_t* p = base + (size_t)row * ld + k + ko;
    FragU f;
    f.u[0] = *(const uint4*)(p);
    f.u[1] = *(const uint4*)(p + 8);
    return f.v;
}

// B fragment for B = (rows of base)^T : 32x16. Lane L -> column (L&15) = row of base;
// lanes 0-15 hold K 0..15 contiguous, lanes 16-31 hold K 16..31.
__device__ inline v16bf loadFragBT(const bf16_t* __restrict__ base, int ld, int k, int lane) {
    const int row = lane & 15;
    const int ko  = (lane >> 4) * 16;
    const bf16_t* p = base + (size_t)row * ld + k + ko;
    FragU f;
    f.u[0] = *(const uint4*)(p);
    f.u[1] = *(const uint4*)(p + 8);
    return f.v;
}

__device__ inline v8f wmma_bf16(v16bf a, v16bf b, v8f c) {
    return __builtin_amdgcn_wmma_f32_16x16x32_bf16(false, a, false, b, (short)0, c, false, false);
}

// ---------------- Kernel 1: f32 -> bf16 + per-channel mean -------------------
__global__ void k_convert(const float* __restrict__ x, bf16_t* __restrict__ qb,
                          float* __restrict__ qmean) {
    const int row = blockIdx.x;             // b*CC + c
    const int tid = threadIdx.x;
    const float* xr = x + (size_t)row * NN;
    bf16_t* qr = qb + (size_t)row * NN;
    float s = 0.f;
#pragma unroll
    for (int j = 0; j < 4; ++j) {
        const int idx = (tid + j * 256) * 4;
        float4 v = *(const float4*)(xr + idx);
        s += v.x + v.y + v.z + v.w;
        union { bf16_t h[4]; uint2 u; } pk;
        pk.h[0] = (bf16_t)v.x; pk.h[1] = (bf16_t)v.y;
        pk.h[2] = (bf16_t)v.z; pk.h[3] = (bf16_t)v.w;
        *(uint2*)(qr + idx) = pk.u;
    }
    __shared__ float red[256];
    red[tid] = s;
    __syncthreads();
    for (int off = 128; off > 0; off >>= 1) {
        if (tid < off) red[tid] += red[tid + off];
        __syncthreads();
    }
    if (tid == 0) qmean[row] = red[0] * (1.0f / NN);
}

// ---------------- Kernel 2: energy = q * q^T (bf16 WMMA, f32 acc) ------------
// Block tile 256(M) x 64(N), 8 waves = 4(M) x 2(N); each wave 64x32 -> 8 accums.
// Per K=32 step: 12 b128 loads feed 8 WMMAs (B frags reused across 4 A frags).
__global__ void k_gemm1(const bf16_t* __restrict__ qb, float* __restrict__ energy) {
    const int b     = blockIdx.z;
    const int mBase = blockIdx.y * 256;
    const int nBase = blockIdx.x * 64;
    const int tid   = threadIdx.x;
    const int lane  = tid & 31;
    const int wave  = tid >> 5;
    const int m0    = mBase + (wave & 3) * 64;
    const int n0    = nBase + (wave >> 2) * 32;

    const bf16_t* qBatch = qb + (size_t)b * CC * NN;
    const bf16_t* pA0 = qBatch + (size_t)m0 * NN;
    const bf16_t* pA1 = pA0 + (size_t)16 * NN;
    const bf16_t* pA2 = pA0 + (size_t)32 * NN;
    const bf16_t* pA3 = pA0 + (size_t)48 * NN;
    const bf16_t* pB0 = qBatch + (size_t)n0 * NN;
    const bf16_t* pB1 = pB0 + (size_t)16 * NN;

    v8f c00 = {}; v8f c01 = {};
    v8f c10 = {}; v8f c11 = {};
    v8f c20 = {}; v8f c21 = {};
    v8f c30 = {}; v8f c31 = {};
    for (int k = 0; k < NN; k += 32) {
        v16bf b0 = loadFragBT(pB0, NN, k, lane);
        v16bf b1 = loadFragBT(pB1, NN, k, lane);
        v16bf a0 = loadFragA (pA0, NN, k, lane);
        v16bf a1 = loadFragA (pA1, NN, k, lane);
        v16bf a2 = loadFragA (pA2, NN, k, lane);
        v16bf a3 = loadFragA (pA3, NN, k, lane);
        c00 = wmma_bf16(a0, b0, c00);
        c01 = wmma_bf16(a0, b1, c01);
        c10 = wmma_bf16(a1, b0, c10);
        c11 = wmma_bf16(a1, b1, c11);
        c20 = wmma_bf16(a2, b0, c20);
        c21 = wmma_bf16(a2, b1, c21);
        c30 = wmma_bf16(a3, b0, c30);
        c31 = wmma_bf16(a3, b1, c31);
    }
    float* e = energy + (size_t)b * CC * CC;
    const int n  = lane & 15;
    const int mh = (lane >> 4) * 8;
#pragma unroll
    for (int r = 0; r < 8; ++r) {
        e[(size_t)(m0 + mh + r)      * CC + (n0 + n)]      = c00[r];
        e[(size_t)(m0 + mh + r)      * CC + (n0 + 16 + n)] = c01[r];
        e[(size_t)(m0 + 16 + mh + r) * CC + (n0 + n)]      = c10[r];
        e[(size_t)(m0 + 16 + mh + r) * CC + (n0 + 16 + n)] = c11[r];
        e[(size_t)(m0 + 32 + mh + r) * CC + (n0 + n)]      = c20[r];
        e[(size_t)(m0 + 32 + mh + r) * CC + (n0 + 16 + n)] = c21[r];
        e[(size_t)(m0 + 48 + mh + r) * CC + (n0 + n)]      = c30[r];
        e[(size_t)(m0 + 48 + mh + r) * CC + (n0 + 16 + n)] = c31[r];
    }
}

// ---------------- Kernel 3: att = softmax(-energy) row-wise, emit bf16 -------
__global__ void k_softmax(const float* __restrict__ energy, bf16_t* __restrict__ att) {
    const int row = blockIdx.x;             // b*CC + c
    const int tid = threadIdx.x;
    const float* e = energy + (size_t)row * CC;
    bf16_t* a = att + (size_t)row * CC;
    const float v0 = -e[tid];
    const float v1 = -e[tid + 256];
    __shared__ float red[256];
    red[tid] = fmaxf(v0, v1);
    __syncthreads();
    for (int off = 128; off > 0; off >>= 1) {
        if (tid < off) red[tid] = fmaxf(red[tid], red[tid + off]);
        __syncthreads();
    }
    const float m = red[0];
    __syncthreads();
    const float p0 = __expf(v0 - m);
    const float p1 = __expf(v1 - m);
    red[tid] = p0 + p1;
    __syncthreads();
    for (int off = 128; off > 0; off >>= 1) {
        if (tid < off) red[tid] += red[tid + off];
        __syncthreads();
    }
    const float inv = 1.0f / red[0];
    a[tid]       = (bf16_t)(p0 * inv);
    a[tid + 256] = (bf16_t)(p1 * inv);
}

// ---------------- Kernel 4: out = att * q (LDS-transposed B tile) ------------
// Block tile 256(M) x 64(N), 8 waves = 4(M) x 2(N); each wave 64x32 -> 8 accums.
__global__ void k_gemm2(const bf16_t* __restrict__ att, const bf16_t* __restrict__ qb,
                        float* __restrict__ out) {
    __shared__ bf16_t ldsB[64][48];         // [n][k], k<32, row = 96 B keeps 16B alignment
    const int b     = blockIdx.z;
    const int mBase = blockIdx.y * 256;
    const int nBase = blockIdx.x * 64;
    const int tid   = threadIdx.x;
    const int lane  = tid & 31;
    const int wave  = tid >> 5;
    const int m0    = mBase + (wave & 3) * 64;
    const int nW    = (wave >> 2) * 32;

    const bf16_t* pA0    = att + (size_t)b * CC * CC + (size_t)m0 * CC;
    const bf16_t* pA1    = pA0 + (size_t)16 * CC;
    const bf16_t* pA2    = pA0 + (size_t)32 * CC;
    const bf16_t* pA3    = pA0 + (size_t)48 * CC;
    const bf16_t* qBatch = qb  + (size_t)b * CC * NN;
    const int kk = tid >> 3;                // 0..31 : K row of staged tile
    const int ng = (tid & 7) * 8;           // 0..56 : N group of 8

    v8f c00 = {}; v8f c01 = {};
    v8f c10 = {}; v8f c11 = {};
    v8f c20 = {}; v8f c21 = {};
    v8f c30 = {}; v8f c31 = {};
    for (int k0 = 0; k0 < CC; k0 += 32) {
        __syncthreads();                    // previous reads done before restage
        union { uint4 u; bf16_t h[8]; } st;
        st.u = *(const uint4*)(qBatch + (size_t)(k0 + kk) * NN + nBase + ng);
#pragma unroll
        for (int i = 0; i < 8; ++i) ldsB[ng + i][kk] = st.h[i];
        __syncthreads();

        const int nr = lane & 15;
        const int ko = (lane >> 4) * 16;
        FragU f0, f1;
        f0.u[0] = *(const uint4*)(&ldsB[nW + nr][ko]);
        f0.u[1] = *(const uint4*)(&ldsB[nW + nr][ko + 8]);
        f1.u[0] = *(const uint4*)(&ldsB[nW + 16 + nr][ko]);
        f1.u[1] = *(const uint4*)(&ldsB[nW + 16 + nr][ko + 8]);
        v16bf a0 = loadFragA(pA0, CC, k0, lane);
        v16bf a1 = loadFragA(pA1, CC, k0, lane);
        v16bf a2 = loadFragA(pA2, CC, k0, lane);
        v16bf a3 = loadFragA(pA3, CC, k0, lane);
        c00 = wmma_bf16(a0, f0.v, c00);
        c01 = wmma_bf16(a0, f1.v, c01);
        c10 = wmma_bf16(a1, f0.v, c10);
        c11 = wmma_bf16(a1, f1.v, c11);
        c20 = wmma_bf16(a2, f0.v, c20);
        c21 = wmma_bf16(a2, f1.v, c21);
        c30 = wmma_bf16(a3, f0.v, c30);
        c31 = wmma_bf16(a3, f1.v, c31);
    }
    float* o = out + (size_t)b * CC * NN;
    const int n  = lane & 15;
    const int mh = (lane >> 4) * 8;
#pragma unroll
    for (int r = 0; r < 8; ++r) {
        o[(size_t)(m0 + mh + r)      * NN + (nBase + nW + n)]      = c00[r];
        o[(size_t)(m0 + mh + r)      * NN + (nBase + nW + 16 + n)] = c01[r];
        o[(size_t)(m0 + 16 + mh + r) * NN + (nBase + nW + n)]      = c10[r];
        o[(size_t)(m0 + 16 + mh + r) * NN + (nBase + nW + 16 + n)] = c11[r];
        o[(size_t)(m0 + 32 + mh + r) * NN + (nBase + nW + n)]      = c20[r];
        o[(size_t)(m0 + 32 + mh + r) * NN + (nBase + nW + 16 + n)] = c21[r];
        o[(size_t)(m0 + 48 + mh + r) * NN + (nBase + nW + n)]      = c30[r];
        o[(size_t)(m0 + 48 + mh + r) * NN + (nBase + nW + 16 + n)] = c31[r];
    }
}

// ---------------- Kernel 5: SE path. mean_n(out) = att * mean_n(q) -----------
__global__ void k_se(const bf16_t* __restrict__ att, const float* __restrict__ qmean,
                     const float* __restrict__ w1, const float* __restrict__ b1,
                     const float* __restrict__ w2, const float* __restrict__ b2,
                     float* __restrict__ se) {
    const int b   = blockIdx.x;
    const int tid = threadIdx.x;
    __shared__ float pooled[2 * CC];
    __shared__ float hbuf[64];
    const float* qm = qmean + b * CC;
    pooled[tid]       = qm[tid];
    pooled[tid + 256] = qm[tid + 256];
    __syncthreads();
#pragma unroll
    for (int rr = 0; rr < 2; ++rr) {
        const int c = tid + rr * 256;
        const bf16_t* ar = att + (size_t)b * CC * CC + (size_t)c * CC;
        float s = 0.f;
        for (int d = 0; d < CC; ++d) s += (float)ar[d] * pooled[d];
        pooled[CC + c] = s;                 // distinct region from pooled[0..CC)
    }
    __syncthreads();
    if (tid < 64) {
        float s = b1[tid];
        const float* wr = w1 + tid * 2 * CC;
        for (int i = 0; i < 2 * CC; ++i) s += wr[i] * pooled[i];
        hbuf[tid] = fmaxf(s, 0.f);
    }
    __syncthreads();
#pragma unroll
    for (int rr = 0; rr < 2; ++rr) {
        const int c = tid + rr * 256;
        float s = b2[c];
        const float* wr = w2 + c * 64;
#pragma unroll
        for (int j = 0; j < 64; ++j) s += wr[j] * hbuf[j];
        se[b * CC + c] = 1.0f / (1.0f + __expf(-s));
    }
}

// ---------------- Kernel 6: result = se*x + (1-se)*out (in-place on d_out) ---
__global__ void k_blend(const float* __restrict__ x, const float* __restrict__ se,
                        float* __restrict__ out) {
    const size_t i4 = (size_t)blockIdx.x * 256 + threadIdx.x;
    const size_t i  = i4 * 4;
    const int bc = (int)(i >> 12);          // NN = 4096 elements per (b,c)
    const float s = se[bc];
    float4 xv = *(const float4*)(x + i);
    float4 ov = *(const float4*)(out + i);
    float4 r;
    r.x = s * xv.x + (1.f - s) * ov.x;
    r.y = s * xv.y + (1.f - s) * ov.y;
    r.z = s * xv.z + (1.f - s) * ov.z;
    r.w = s * xv.w + (1.f - s) * ov.w;
    *(float4*)(out + i) = r;
}

extern "C" void kernel_launch(void* const* d_in, const int* in_sizes, int n_in,
                              void* d_out, int out_size, void* d_ws, size_t ws_size,
                              hipStream_t stream) {
    (void)in_sizes; (void)n_in; (void)out_size; (void)ws_size;
    const float* x  = (const float*)d_in[0];
    const float* w1 = (const float*)d_in[1];
    const float* b1 = (const float*)d_in[2];
    const float* w2 = (const float*)d_in[3];
    const float* b2 = (const float*)d_in[4];
    float* out = (float*)d_out;

    char* ws = (char*)d_ws;
    // layout: qb (bf16, 64 MiB) | energy (f32, 16 MiB) | att (bf16, 8 MiB) | qmean | se
    bf16_t* qb     = (bf16_t*)(ws);
    float*  energy = (float*) (ws + (size_t)67108864);
    bf16_t* att    = (bf16_t*)(ws + (size_t)67108864 + 16777216);
    float*  qmean  = (float*) (ws + (size_t)67108864 + 16777216 + 8388608);
    float*  se     = (float*) (ws + (size_t)67108864 + 16777216 + 8388608 + 32768);

    k_convert<<<dim3(BB * CC), 256, 0, stream>>>(x, qb, qmean);
    k_gemm1  <<<dim3(CC / 64, CC / 256, BB), 256, 0, stream>>>(qb, energy);
    k_softmax<<<dim3(BB * CC), 256, 0, stream>>>(energy, att);
    k_gemm2  <<<dim3(NN / 64, CC / 256, BB), 256, 0, stream>>>(att, qb, out);
    k_se     <<<dim3(BB), 256, 0, stream>>>(att, qmean, w1, b1, w2, b2, se);
    k_blend  <<<dim3(32768), 256, 0, stream>>>(x, se, out);
}
